// Model_24910810317301
// MI455X (gfx1250) — matrix operand order repaired and checked by reading, not compiled, not run on hardware
//
#include <hip/hip_runtime.h>
#include <hip/hip_bf16.h>
#include <cmath>
#include <cstdint>

// ---------------- CDNA5 WMMA / TDM types ----------------
typedef __attribute__((ext_vector_type(16))) __bf16 v16bf;
typedef __attribute__((ext_vector_type(8)))  float  v8f;
typedef __attribute__((ext_vector_type(4)))  unsigned u32x4;
typedef __attribute__((ext_vector_type(8)))  int      i32x8;
typedef __attribute__((ext_vector_type(4)))  int      i32x4;

#define SEQLEN 5120
#define BSTRIDE 4864
#define WCT 163840L   // per-window activation stride in elements (max C*T = 32*5120)

__device__ __forceinline__ unsigned short f2bf(float f) {
    unsigned u = __float_as_uint(f);
    unsigned r = (u + 0x7FFFu + ((u >> 16) & 1u)) >> 16;   // RNE to bf16
    return (unsigned short)r;
}
__device__ __forceinline__ float bf2f(unsigned short h) {
    return __uint_as_float((unsigned)h << 16);
}

// =====================================================================
// 0) Weight pre-pack: fp32 [Cout][Ktot] -> bf16 A-fragments in lane order.
//    (this code's contiguous k order, not the ISA table order; 16-bit A 16x32): lane l row m=l&15; elem j:
//    k = 16*(l>>4)+j. 32B contiguous per lane.
// =====================================================================
__global__ __launch_bounds__(32)
void pack_w_kernel(const float* __restrict__ W, unsigned short* __restrict__ Wp,
                   int Cout, int Ktot, int nk)
{
    const int kb = blockIdx.x, mt = blockIdx.y, lane = threadIdx.x;
    const int m = mt * 16 + (lane & 15);
    const int half = lane >> 4;
    unsigned short o[16];
#pragma unroll
    for (int j = 0; j < 16; ++j) {
        int k = kb * 32 + 16 * half + j;
        float w = (m < Cout && k < Ktot) ? W[(long)m * Ktot + k] : 0.f;
        o[j] = f2bf(w);
    }
    unsigned short* dst = Wp + ((long)(mt * nk + kb) * 32 + lane) * 16;
    unsigned w32[8];
#pragma unroll
    for (int q = 0; q < 8; ++q) w32[q] = (unsigned)o[2 * q] | ((unsigned)o[2 * q + 1] << 16);
    *(uint4*)dst       = make_uint4(w32[0], w32[1], w32[2], w32[3]);
    *(uint4*)(dst + 8) = make_uint4(w32[4], w32[5], w32[6], w32[7]);
}

// =====================================================================
// 1) Window gather + mean-subtract + abs-max normalize -> bf16
// =====================================================================
__global__ __launch_bounds__(256)
void prep_kernel(const float* __restrict__ x, unsigned short* __restrict__ wave, int Tx)
{
    __shared__ float red[256];
    __shared__ float sMean, sScale;
    const int b = blockIdx.z, c = blockIdx.y, tid = threadIdx.x;
    const long base = (long)b * BSTRIDE;

    float sum = 0.f;
    for (int i = tid; i < SEQLEN; i += 256) {
        long t = base + i; if (t > Tx - 1) t = Tx - 1;
        sum += x[t * 3 + c];
    }
    red[tid] = sum; __syncthreads();
    for (int off = 128; off > 0; off >>= 1) {
        if (tid < off) red[tid] += red[tid + off];
        __syncthreads();
    }
    if (tid == 0) sMean = red[0] * (1.f / SEQLEN);
    __syncthreads();
    const float mean = sMean;

    float mx = 0.f;
    for (int i = tid; i < SEQLEN; i += 256) {
        long t = base + i; if (t > Tx - 1) t = Tx - 1;
        mx = fmaxf(mx, fabsf(x[t * 3 + c] - mean));
    }
    red[tid] = mx; __syncthreads();
    for (int off = 128; off > 0; off >>= 1) {
        if (tid < off) red[tid] = fmaxf(red[tid], red[tid + off]);
        __syncthreads();
    }
    if (tid == 0) sScale = 1.f / (red[0] + 1e-6f);
    __syncthreads();
    const float sc = sScale;

    unsigned short* wv = wave + (long)b * 15360 + (long)c * SEQLEN;
    for (int i = tid; i < SEQLEN; i += 256) {
        long t = base + i; if (t > Tx - 1) t = Tx - 1;
        wv[i] = f2bf((x[t * 3 + c] - mean) * sc);
    }
}

// =====================================================================
// 2) Conv-as-GEMM on WMMA.
//    Block = 128 threads (4 waves) -> 16(Cout) x 128(T) output tile.
//    Ks==1: TDM path. Wave0 issues TENSOR_LOAD_TO_LDS for the 32Kx128T
//    bf16 tile (D# per ISA ch.8; OOB rows/cols hardware-zero-filled),
//    double-buffered across K-steps with s_wait_tensorcnt. Compute waves
//    gather B fragments from the row-major LDS tile.
//    Ks==5: im2col shuffle-on-write staging (bf16) + ds_load_b128 frags.
//    2x v_wmma_f32_16x16x32_bf16 per wave per K-step.
//    Fused bias->BN->ReLU->residual epilogue, bf16 out.
// =====================================================================
__global__ __launch_bounds__(128)
void conv_wmma_kernel(const unsigned short* __restrict__ X,   // [B][Cin][Tin] bf16
                      const unsigned short* __restrict__ Wp,  // packed bf16 fragments
                      const float* __restrict__ bias,
                      const float* __restrict__ bng, const float* __restrict__ bnb,
                      const float* __restrict__ bnm, const float* __restrict__ bnv,
                      const unsigned short* __restrict__ Res, // bf16 or null
                      unsigned short* __restrict__ Y,         // bf16
                      int Cin, int Cout, int Tin, int Ks, int pad, int nk,
                      long xsB, long ysB, long rsB, int relu)
{
    __shared__ __align__(32) unsigned short stage[2][4096];   // 2 x 8KB [32 K][128 T]
    const int tid    = threadIdx.x;
    const int lane   = tid & 31;
    const int wv     = tid >> 5;
    const int half   = lane >> 4;
    const int lane16 = lane & 15;
    const int tTile  = blockIdx.x * 128;
    const int mt     = blockIdx.y;
    const int b      = blockIdx.z;

    const unsigned short* Xb = X + (long)b * xsB;
    const int Ktot = Cin * Ks;
    const int n0   = tTile + (2 * wv) * 16 + lane16;
    const int n1   = n0 + 16;

    v8f acc0 = {}, acc1 = {};

    if (Ks == 1) {
        // -------- TDM double-buffered path --------
        auto issue = [&](int kb, int bufi) {
            unsigned lds = (unsigned)(uintptr_t)(&stage[bufi][0]);
            unsigned long long ga =
                (unsigned long long)(uintptr_t)(Xb + (long)(kb * 32) * Tin + tTile);
            unsigned td0 = (unsigned)(Tin - tTile);     // remaining cols (OOB -> 0)
            unsigned td1 = (unsigned)(Cin - kb * 32);   // remaining rows (OOB -> 0)
            u32x4 g0;
            g0[0] = 1u;                                        // count=1, user mode
            g0[1] = lds;                                       // lds_addr
            g0[2] = (unsigned)(ga & 0xFFFFFFFFu);              // global_addr[31:0]
            g0[3] = (unsigned)((ga >> 32) & 0x1FFFFFFu) | (2u << 30); // [56:32] | type=2
            i32x8 g1;
            g1[0] = (int)(1u << 16);                           // data_size = 2B
            g1[1] = (int)((td0 & 0xFFFFu) << 16);              // tensor_dim0[15:0]
            g1[2] = (int)((td0 >> 16) | ((td1 & 0xFFFFu) << 16)); // dim0[31:16]|dim1[15:0]
            g1[3] = (int)((td1 >> 16) | (128u << 16));         // dim1[31:16]|tile_dim0=128
            g1[4] = (int)32u;                                  // tile_dim1=32, tile_dim2=0
            g1[5] = (int)(unsigned)Tin;                        // tensor_dim0_stride[31:0]
            g1[6] = 0; g1[7] = 0;
            i32x4 g2; g2[0] = 1; g2[1] = 1; g2[2] = 0; g2[3] = 0; // dim2=dim3=1
            i32x4 g3; g3[0] = 0; g3[1] = (int)(1u << 16); g3[2] = 0; g3[3] = 0; // dim4=1
            i32x8 g4 = {};                                     // VADDR4 unused
            __builtin_amdgcn_tensor_load_to_lds(g0, g1, g2, g3, g4, 0);
        };
        if (wv == 0) issue(0, 0);
        for (int kb = 0; kb < nk; ++kb) {
            if (wv == 0) {
                if (kb + 1 < nk) {
                    issue(kb + 1, (kb + 1) & 1);
                    __builtin_amdgcn_s_wait_tensorcnt(1);   // tile kb landed
                } else {
                    __builtin_amdgcn_s_wait_tensorcnt(0);
                }
            }
            __syncthreads();
            const unsigned short* st = stage[kb & 1];
            v16bf afrag = *(const v16bf*)(Wp + ((long)(mt * nk + kb) * 32 + lane) * 16);
            union { v16bf v; unsigned short s[16]; } fb0, fb1;
#pragma unroll
            for (int j = 0; j < 16; ++j) {
                int row = 16 * half + j;                    // K within step
                fb0.s[j] = st[row * 128 + (2 * wv) * 16 + lane16];
                fb1.s[j] = st[row * 128 + (2 * wv) * 16 + 16 + lane16];
            }
            acc0 = __builtin_amdgcn_wmma_f32_16x16x32_bf16(
                       false, afrag, false, fb0.v, (short)0, acc0, false, false);
            acc1 = __builtin_amdgcn_wmma_f32_16x16x32_bf16(
                       false, afrag, false, fb1.v, (short)0, acc1, false, false);
            __syncthreads();
        }
    } else {
        // -------- im2col shuffle-on-write path (k=5 convs) --------
        unsigned* st32 = (unsigned*)stage[0];
        for (int kb = 0; kb < nk; ++kb) {
            const int k0 = kb * 32;
#pragma unroll
            for (int it = 0; it < 16; ++it) {
                int p  = tid + it * 128;
                int k2 = (p >> 7) << 1;
                int t  = p & 127;
                int n  = tTile + t;
                unsigned short h0 = 0, h1 = 0;
                if (n < Tin) {
                    int kl0 = k0 + k2, kl1 = kl0 + 1;
                    if (kl0 < Ktot) {
                        int c = kl0 / Ks; int tt = n + (kl0 - c * Ks) - pad;
                        if (tt >= 0 && tt < Tin) h0 = Xb[(long)c * Tin + tt];
                    }
                    if (kl1 < Ktot) {
                        int c = kl1 / Ks; int tt = n + (kl1 - c * Ks) - pad;
                        if (tt >= 0 && tt < Tin) h1 = Xb[(long)c * Tin + tt];
                    }
                }
                int nt = t >> 4, slane = (t & 15) + 16 * (k2 >> 4);
                st32[nt * 256 + slane * 8 + ((k2 & 15) >> 1)] =
                    (unsigned)h0 | ((unsigned)h1 << 16);
            }
            __syncthreads();
            v16bf afrag = *(const v16bf*)(Wp + ((long)(mt * nk + kb) * 32 + lane) * 16);
            v16bf bf0 = *(const v16bf*)(st32 + (2 * wv) * 256 + lane * 8);
            v16bf bf1 = *(const v16bf*)(st32 + (2 * wv + 1) * 256 + lane * 8);
            acc0 = __builtin_amdgcn_wmma_f32_16x16x32_bf16(
                       false, afrag, false, bf0, (short)0, acc0, false, false);
            acc1 = __builtin_amdgcn_wmma_f32_16x16x32_bf16(
                       false, afrag, false, bf1, (short)0, acc1, false, false);
            __syncthreads();
        }
    }

    // ---- fused epilogue: (+bias) -> BN -> ReLU -> (+residual) -> bf16 ----
    unsigned short* Yb = Y + (long)b * ysB;
#pragma unroll
    for (int r = 0; r < 8; ++r) {
        int m = mt * 16 + 8 * half + r;
        if (m < Cout) {
            float inv = bng[m] * rsqrtf(bnv[m] + 1e-5f);
            float sh  = bnb[m] - bnm[m] * inv;
            float bi  = bias ? bias[m] : 0.f;
            if (n0 < Tin) {
                float v = (acc0[r] + bi) * inv + sh;
                if (relu) v = fmaxf(v, 0.f);
                if (Res)  v += bf2f(Res[(long)b * rsB + (long)m * Tin + n0]);
                Yb[(long)m * Tin + n0] = f2bf(v);
            }
            if (n1 < Tin) {
                float v = (acc1[r] + bi) * inv + sh;
                if (relu) v = fmaxf(v, 0.f);
                if (Res)  v += bf2f(Res[(long)b * rsB + (long)m * Tin + n1]);
                Yb[(long)m * Tin + n1] = f2bf(v);
            }
        }
    }
}

// =====================================================================
// 3) Depthwise 5-tap conv (pad=2, stride s) + BN + ReLU  (bf16 io)
// =====================================================================
__global__ __launch_bounds__(256)
void dw_conv_kernel(const unsigned short* __restrict__ X, const float* __restrict__ W5,
                    const float* __restrict__ g, const float* __restrict__ bb,
                    const float* __restrict__ m, const float* __restrict__ v,
                    unsigned short* __restrict__ Y, int Tin, int Tout, int stride,
                    long xsB, long ysB)
{
    const int to = blockIdx.x * 256 + threadIdx.x;
    if (to >= Tout) return;
    const int c = blockIdx.y, b = blockIdx.z;
    const unsigned short* xr = X + (long)b * xsB + (long)c * Tin;
    const int t0 = to * stride - 2;
    float acc = 0.f;
#pragma unroll
    for (int k = 0; k < 5; ++k) {
        int t = t0 + k;
        if (t >= 0 && t < Tin) acc += bf2f(xr[t]) * W5[c * 5 + k];
    }
    float inv = g[c] * rsqrtf(v[c] + 1e-5f);
    acc = acc * inv + (bb[c] - m[c] * inv);
    Y[(long)b * ysB + (long)c * Tout + to] = f2bf(fmaxf(acc, 0.f));
}

// =====================================================================
// 4) Bilinear x2 upsample, align_corners=True  (bf16 io)
// =====================================================================
__global__ __launch_bounds__(256)
void up2_kernel(const unsigned short* __restrict__ X, unsigned short* __restrict__ Y,
                int Tin, long xsB, long ysB)
{
    const int Tout = 2 * Tin;
    const int t = blockIdx.x * 256 + threadIdx.x;
    if (t >= Tout) return;
    const int c = blockIdx.y, b = blockIdx.z;
    const float pos = t * ((float)(Tin - 1) / (float)(Tout - 1));
    int i0 = (int)floorf(pos);
    int i1 = i0 + 1; if (i1 > Tin - 1) i1 = Tin - 1;
    float w = pos - (float)i0;
    const unsigned short* xr = X + (long)b * xsB + (long)c * Tin;
    Y[(long)b * ysB + (long)c * Tout + t] = f2bf(bf2f(xr[i0]) * (1.f - w) + bf2f(xr[i1]) * w);
}

// =====================================================================
// 5) Head: 192->7 softmax + time regression; thresholded class scores
// =====================================================================
__global__ __launch_bounds__(128)
void head_kernel(const unsigned short* __restrict__ x1, const unsigned short* __restrict__ h2,
                 const float* __restrict__ clw, const float* __restrict__ clb,
                 const float* __restrict__ tmw, const float* __restrict__ tmb,
                 float* __restrict__ times, float* __restrict__ scores,
                 long sB, int N)
{
    const int t = blockIdx.x * 128 + threadIdx.x;
    if (t >= 640) return;
    const int b = blockIdx.z;
    const unsigned short* xa = x1 + (long)b * sB;
    const unsigned short* xb = h2 + (long)b * sB;

    float lg[7];
#pragma unroll
    for (int c = 0; c < 7; ++c) lg[c] = clb[c];
    float tm = tmb[0];
    for (int f = 0; f < 96; ++f) {
        float va = bf2f(xa[f * 640 + t]);
        float vb = bf2f(xb[f * 640 + t]);
#pragma unroll
        for (int c = 0; c < 7; ++c)
            lg[c] += clw[c * 192 + f] * va + clw[c * 192 + 96 + f] * vb;
        tm += tmw[f] * va + tmw[96 + f] * vb;
    }
    float mx = lg[0];
#pragma unroll
    for (int c = 1; c < 7; ++c) mx = fmaxf(mx, lg[c]);
    float e[7], se = 0.f;
#pragma unroll
    for (int c = 0; c < 7; ++c) { e[c] = __expf(lg[c] - mx); se += e[c]; }
    const float inv = 1.f / se;

    const int idx = b * 640 + t;
    times[idx] = tm * 8.f + (float)t * 8.f + (float)b * (float)BSTRIDE;
#pragma unroll
    for (int c = 0; c < 6; ++c) {
        float p = e[c + 1] * inv;
        scores[(long)c * N + idx] = (p > 0.3f) ? p : -INFINITY;
    }
}

// =====================================================================
// 6) Greedy temporal NMS, one block per class, alive bitmask in LDS
// =====================================================================
__global__ __launch_bounds__(256)
void nms_kernel(const float* __restrict__ scores, const float* __restrict__ times,
                float* __restrict__ out, int N)
{
    __shared__ unsigned alive[4096];
    __shared__ float rs[256];
    __shared__ int   ri[256];
    __shared__ float curT;
    __shared__ int   curKeep;

    const int cls = blockIdx.x, tid = threadIdx.x;
    if (N > 4096 * 32) N = 4096 * 32;
    const int nw = (N + 31) / 32;
    const float* s = scores + (long)cls * N;

    for (int w = tid; w < nw; w += 256) {
        unsigned bits = 0;
        for (int j = 0; j < 32; ++j) {
            int i = w * 32 + j;
            if (i < N && s[i] > -1e30f) bits |= (1u << j);
        }
        alive[w] = bits;
    }
    __syncthreads();

    for (int it = 0; it < 256; ++it) {
        float bestS = -INFINITY; int bestI = -1;
        for (int i = tid; i < N; i += 256) {
            if ((alive[i >> 5] >> (i & 31)) & 1u) {
                float v = s[i];
                if (v > bestS) { bestS = v; bestI = i; }
            }
        }
        rs[tid] = bestS; ri[tid] = bestI; __syncthreads();
        for (int off = 128; off > 0; off >>= 1) {
            if (tid < off && rs[tid + off] > rs[tid]) {
                rs[tid] = rs[tid + off]; ri[tid] = ri[tid + off];
            }
            __syncthreads();
        }
        if (tid == 0) {
            long o = ((long)cls * 256 + it) * 3;
            int i = ri[0];
            if (i >= 0) {
                curKeep = 1; curT = times[i];
                out[o + 0] = (float)cls; out[o + 1] = curT; out[o + 2] = rs[0];
            } else {
                curKeep = 0;
                out[o + 0] = -1.f; out[o + 1] = 0.f; out[o + 2] = 0.f;
            }
        }
        __syncthreads();
        if (curKeep) {
            const float ti = curT;
            for (int w = tid; w < nw; w += 256) {
                unsigned bits = alive[w];
                if (!bits) continue;
                for (int j = 0; j < 32; ++j) {
                    if ((bits >> j) & 1u) {
                        int i = w * 32 + j;
                        if (fabsf(times[i] - ti) <= 500.f) bits &= ~(1u << j);
                    }
                }
                alive[w] = bits;
            }
        }
        __syncthreads();
    }
}

// =====================================================================
// Host orchestration
// Param pytree (sorted-key flatten), x at d_in[0], params at d_in[1..193]:
//   1 cl_b | 2 cl_w | 3+15i cls_ir[i]{dw_bn b,g,m,v; dw_w; pj_bn b,g,m,v; pj_w; pw_bn b,g,m,v; pw_w}
//   63+6i cls_up[i]{b; bn b,g,m,v; w} | 87+15i layers[i] (same as cls_ir) | 192 tm_b | 193 tm_w
// =====================================================================
struct PackedW { const unsigned short* p; int nk; };

extern "C" void kernel_launch(void* const* d_in, const int* in_sizes, int n_in,
                              void* d_out, int out_size, void* d_ws, size_t ws_size,
                              hipStream_t stream)
{
    (void)n_in; (void)out_size; (void)ws_size;
    const float* x = (const float*)d_in[0];
    const int Tx = in_sizes[0] / 3;
    const int B  = (Tx + BSTRIDE - 1) / BSTRIDE;
    auto F = [&](int i) { return (const float*)d_in[i]; };

    // ---- workspace carve-up (bytes, 32B aligned) ----
    char* wsb = (char*)d_ws;
    size_t off = 0;
    auto alloc = [&](size_t bytes) -> void* {
        off = (off + 31) & ~(size_t)31;
        void* p = wsb + off; off += bytes; return p;
    };
    unsigned short* wave = (unsigned short*)alloc((size_t)B * 15360 * 2);
    unsigned short* sb[3];
    for (int i = 0; i < 3; ++i) sb[i] = (unsigned short*)alloc((size_t)B * WCT * 2);
    const int N = B * 640;
    float* times  = (float*)alloc((size_t)N * 4);
    float* scores = (float*)alloc((size_t)6 * N * 4);
    unsigned short* wp_base = (unsigned short*)alloc((size_t)8 << 20);  // packed weights
    long woff = 0;

    auto packW = [&](const float* W, int Cout, int Ktot) -> PackedW {
        int nk = (Ktot + 31) / 32, nmt = (Cout + 15) / 16;
        unsigned short* p = wp_base + woff;
        woff += (long)nmt * nk * 512;
        pack_w_kernel<<<dim3(nk, nmt, 1), 32, 0, stream>>>(W, p, Cout, Ktot, nk);
        return {p, nk};
    };

    // ---- 0) windows + normalize ----
    prep_kernel<<<dim3(1, 3, B), 256, 0, stream>>>(x, wave, Tx);

    // ---- 1) backbone: 7 inverted-residual blocks ----
    static const int L_ci[7] = {3, 8, 16, 16, 32, 32, 64};
    static const int L_co[7] = {8, 16, 16, 32, 32, 64, 96};
    static const int L_s [7] = {1, 1, 2, 1, 2, 1, 2};
    unsigned short* cur = wave; long curS = 15360; int curT = SEQLEN; int ci_idx = -1;
    for (int i = 0; i < 7; ++i) {
        const int base = 87 + 15 * i;
        const int ci = L_ci[i], co = L_co[i], st = L_s[i], h = 2 * ci;
        const int P = (ci_idx + 1) % 3, Q = (ci_idx + 2) % 3;
        const int Tout = (curT - 1) / st + 1;
        PackedW wpw = packW(F(base + 14), h, ci);
        PackedW wpj = packW(F(base + 9), co, h);
        conv_wmma_kernel<<<dim3((curT + 127) / 128, (h + 15) / 16, B), 128, 0, stream>>>(
            cur, wpw.p, nullptr,
            F(base + 11), F(base + 10), F(base + 12), F(base + 13),
            nullptr, sb[P], ci, h, curT, 1, 0, wpw.nk, curS, WCT, 0, 1);
        dw_conv_kernel<<<dim3((Tout + 255) / 256, h, B), 256, 0, stream>>>(
            sb[P], F(base + 4),
            F(base + 1), F(base + 0), F(base + 2), F(base + 3),
            sb[Q], curT, Tout, st, WCT, WCT);
        const unsigned short* res = (st == 1 && ci == co) ? cur : nullptr;
        conv_wmma_kernel<<<dim3((Tout + 127) / 128, (co + 15) / 16, B), 128, 0, stream>>>(
            sb[Q], wpj.p, nullptr,
            F(base + 6), F(base + 5), F(base + 7), F(base + 8),
            res, sb[P], h, co, Tout, 1, 0, wpj.nk, WCT, WCT, curS, 0);
        cur = sb[P]; curS = WCT; curT = Tout; ci_idx = P;
    }
    unsigned short* x1p = cur;   // [96, 640] per window, preserved below

    // ---- 2) cls down branch: 4 IR blocks, stride 2 ----
    static const int C_ci[4] = {96, 128, 156, 200};
    static const int C_co[4] = {128, 156, 200, 256};
    unsigned short* d1 = sb[(ci_idx + 1) % 3];
    unsigned short* d2 = sb[(ci_idx + 2) % 3];
    unsigned short* cur2 = x1p; int c2T = 640;
    for (int i = 0; i < 4; ++i) {
        const int base = 3 + 15 * i;
        const int ci = C_ci[i], co = C_co[i], h = 2 * ci;
        const int Tout = (c2T - 1) / 2 + 1;
        PackedW wpw = packW(F(base + 14), h, ci);
        PackedW wpj = packW(F(base + 9), co, h);
        conv_wmma_kernel<<<dim3((c2T + 127) / 128, (h + 15) / 16, B), 128, 0, stream>>>(
            cur2, wpw.p, nullptr,
            F(base + 11), F(base + 10), F(base + 12), F(base + 13),
            nullptr, d1, ci, h, c2T, 1, 0, wpw.nk, WCT, WCT, 0, 1);
        dw_conv_kernel<<<dim3((Tout + 255) / 256, h, B), 256, 0, stream>>>(
            d1, F(base + 4),
            F(base + 1), F(base + 0), F(base + 2), F(base + 3),
            d2, c2T, Tout, 2, WCT, WCT);
        conv_wmma_kernel<<<dim3((Tout + 127) / 128, (co + 15) / 16, B), 128, 0, stream>>>(
            d2, wpj.p, nullptr,
            F(base + 6), F(base + 5), F(base + 7), F(base + 8),
            nullptr, d1, h, co, Tout, 1, 0, wpj.nk, WCT, WCT, 0, 0);
        cur2 = d1; c2T = Tout;
        unsigned short* t = d1; d1 = d2; d2 = t;
    }

    // ---- 3) cls up branch: 4x (up2 -> conv k=5 + bias -> BN -> ReLU) ----
    static const int U_ci[4] = {256, 200, 156, 128};
    static const int U_co[4] = {200, 156, 128, 96};
    unsigned short* freeb = d1;
    for (int i = 0; i < 4; ++i) {
        const int base = 63 + 6 * i;
        const int ci = U_ci[i], co = U_co[i];
        const int Tout = 2 * c2T;
        PackedW wpu = packW(F(base + 5), co, ci * 5);
        up2_kernel<<<dim3((Tout + 255) / 256, ci, B), 256, 0, stream>>>(
            cur2, freeb, c2T, WCT, WCT);
        unsigned short* dst = cur2;         // old cur2 buffer, consumed by up2
        conv_wmma_kernel<<<dim3((Tout + 127) / 128, (co + 15) / 16, B), 128, 0, stream>>>(
            freeb, wpu.p, F(base + 0),
            F(base + 2), F(base + 1), F(base + 3), F(base + 4),
            nullptr, dst, ci, co, Tout, 5, 2, wpu.nk, WCT, WCT, 0, 1);
        cur2 = dst; c2T = Tout;
    }

    // ---- 4) head: softmax + times + thresholded class scores ----
    head_kernel<<<dim3((640 + 127) / 128, 1, B), 128, 0, stream>>>(
        x1p, cur2, F(2), F(1), F(193), F(192), times, scores, WCT, N);

    // ---- 5) per-class greedy temporal NMS -> d_out [6*256, 3] ----
    nms_kernel<<<dim3(6, 1, 1), 256, 0, stream>>>(scores, times, (float*)d_out, N);
}
